// Model_43404939493957
// MI455X (gfx1250) — matrix operand order repaired and checked by reading, not compiled, not run on hardware
//
#include <hip/hip_runtime.h>
#include <hip/hip_bf16.h>
#include <math.h>

#define DEV __device__ __forceinline__

typedef __attribute__((ext_vector_type(16))) __bf16 v16bf;
typedef __attribute__((ext_vector_type(8)))  float  v8f;

union Frag16 { unsigned int u[8]; v16bf v; };

static constexpr int Bb   = 16;
static constexpr int Tt   = 4096;
static constexpr int INF  = 64;
static constexpr int Hh   = 512;
static constexpr int Nn   = 256;
static constexpr int Ll   = 4;
static constexpr int OUTF = 64;
static constexpr int G3   = 3 * Nn;            // 768
static constexpr long M_ROWS = (long)Bb * Tt;  // 65536 rows, R = t*16 + b

// scan-kernel LDS geometry (padded strides to kill bank conflicts)
static constexpr int HBS = 264;                // hF/hB row stride (elements)
static constexpr int HGS = 772;                // hg row stride (dwords)
static constexpr int IGSTEP = 16 * G3;         // 12288 floats = 49152 B per scan step
static constexpr int LDS_HF  = 0;
static constexpr int LDS_HB  = 16 * HBS * 4;                   // 16896
static constexpr int LDS_HG  = LDS_HB + 16 * HBS * 2;          // 25344
static constexpr int LDS_IG  = LDS_HG + 16 * HGS * 4;          // 74752 (16B aligned)
static constexpr int LDS_TOT = LDS_IG + 2 * IGSTEP * 4;        // 173056

// ---------- scalar helpers (native bf16 converts -> v_cvt, not ALU emulation) ----------
DEV unsigned short f2bf(float f) {
  union { __bf16 b; unsigned short s; } c;
  c.b = (__bf16)f;
  return c.s;
}
DEV unsigned int pack2(float lo, float hi) {
  union { __bf16 b[2]; unsigned int u; } c;
  c.b[0] = (__bf16)lo;
  c.b[1] = (__bf16)hi;
  return c.u;
}
DEV float sigmoidf_(float x) { return 1.0f / (1.0f + __expf(-x)); }
DEV float geluf_(float x) {
  float x3 = x * x * x;
  return 0.5f * x * (1.0f + tanhf(0.7978845608028654f * (x + 0.044715f * x3)));
}
DEV v8f v8zero() { v8f z = {0.f,0.f,0.f,0.f,0.f,0.f,0.f,0.f}; return z; }

// ---------- WMMA fragment loaders (per CDNA5 ISA VGPR layouts, b128-vectorized) ----------
// bf16 A 16x32: lane holds K = {8h..8h+7} (VGPR0-3) and {16+8h..16+8h+7} (VGPR4-7), h=lane>>4.
// Both spans are 16 contiguous bytes -> two b128 loads.
DEV v16bf load_a_b16(const unsigned short* __restrict__ A, long rowoff, int kbase, int lane) {
  Frag16 f; int half = lane >> 4;
  const uint4 lo = *(const uint4*)(A + rowoff + kbase + 8 * half);
  const uint4 hi = *(const uint4*)(A + rowoff + kbase + 16 + 8 * half);
  f.u[0] = lo.x; f.u[1] = lo.y; f.u[2] = lo.z; f.u[3] = lo.w;
  f.u[4] = hi.x; f.u[5] = hi.y; f.u[6] = hi.z; f.u[7] = hi.w;
  return f.v;
}
// fp32 source variant: four float4 loads + packed converts.
DEV v16bf load_a_f32(const float* __restrict__ A, long rowoff, int kbase, int lane) {
  Frag16 f; int half = lane >> 4;
  const float* p0 = A + rowoff + kbase + 8 * half;        // K = 8h..8h+7
  const float* p1 = A + rowoff + kbase + 16 + 8 * half;   // K = 16+8h..+7
  float4 a0 = *(const float4*)p0, a1 = *(const float4*)(p0 + 4);
  float4 b0 = *(const float4*)p1, b1 = *(const float4*)(p1 + 4);
  f.u[0] = pack2(a0.x, a0.y); f.u[1] = pack2(a0.z, a0.w);
  f.u[2] = pack2(a1.x, a1.y); f.u[3] = pack2(a1.z, a1.w);
  f.u[4] = pack2(b0.x, b0.y); f.u[5] = pack2(b0.z, b0.w);
  f.u[6] = pack2(b1.x, b1.y); f.u[7] = pack2(b1.z, b1.w);
  return f.v;
}
// bf16 B 32x16: lane holds K = {8h..8h+7} and {16+8h..16+8h+7} of column n -> two b128 loads.
// B[k][n] = W[n*K + k]; caller passes n = ntile*16 + (lane&15).
DEV v16bf load_b_b16(const unsigned short* __restrict__ W, long n, int K, int kbase, int lane) {
  Frag16 f; int half = lane >> 4;
  const unsigned short* p = W + n * (long)K + kbase + 8 * half;
  const uint4 lo = *(const uint4*)p;
  const uint4 hi = *(const uint4*)(p + 16);
  f.u[0] = lo.x; f.u[1] = lo.y; f.u[2] = lo.z; f.u[3] = lo.w;
  f.u[4] = hi.x; f.u[5] = hi.y; f.u[6] = hi.z; f.u[7] = hi.w;
  return f.v;
}
DEV v8f wmma_bf16(v16bf a, v16bf b, v8f c) {
  return __builtin_amdgcn_wmma_f32_16x16x32_bf16(false, a, false, b, (short)0, c, false, false);
}
// f32 C/D 16x16: VGPR r -> M = r + 8*(lane>>4), N = lane&15

// ---------- fp32 -> bf16 weight conversion ----------
__global__ void k_cvt(const float* __restrict__ s, unsigned short* __restrict__ d, int n) {
  int i = blockIdx.x * 256 + threadIdx.x;
  if (i < n) d[i] = f2bf(s[i]);
}

// ---------- encoder: h[T,B,H] = x[B,T,IN] @ enc_W^T + enc_b ----------
__global__ void k_encoder(const float* __restrict__ x, const unsigned short* __restrict__ W16,
                          const float* __restrict__ bias, float* __restrict__ h) {
  int lane = threadIdx.x & 31, wave = threadIdx.x >> 5;
  int mt = blockIdx.x;                        // == t
  int ntb = (blockIdx.y * 4 + wave) * 4;      // 0,4,...,28
  long arow = ((long)(lane & 15) * Tt + mt) * INF;   // x row (b=lane&15, t=mt)
  v8f acc[4] = {v8zero(), v8zero(), v8zero(), v8zero()};
#pragma unroll
  for (int kc = 0; kc < INF / 32; ++kc) {
    v16bf a = load_a_f32(x, arow, kc * 32, lane);
#pragma unroll
    for (int j = 0; j < 4; ++j) {
      long n = (long)(ntb + j) * 16 + (lane & 15);
      acc[j] = wmma_bf16(a, load_b_b16(W16, n, INF, kc * 32, lane), acc[j]);
    }
  }
  int half = lane >> 4;
#pragma unroll
  for (int j = 0; j < 4; ++j) {
    long n = (long)(ntb + j) * 16 + (lane & 15);
    float bv = bias[n];
#pragma unroll
    for (int r = 0; r < 8; ++r)
      h[((long)mt * 16 + r + 8 * half) * Hh + n] = acc[j][r] + bv;
  }
}

// ---------- layernorm: xn_bf16[R,H] = LN(h[R,H]) * g + b ----------
__global__ void k_layernorm(const float* __restrict__ h, const float* __restrict__ g,
                            const float* __restrict__ b, unsigned short* __restrict__ xn) {
  int lane = threadIdx.x & 31, wave = threadIdx.x >> 5;
  long row = (long)blockIdx.x * 8 + wave;
  const float* hp = h + row * Hh;
  float v[16]; float s = 0.f, s2 = 0.f;
#pragma unroll
  for (int i = 0; i < 16; ++i) { v[i] = hp[lane + i * 32]; s += v[i]; s2 += v[i] * v[i]; }
#pragma unroll
  for (int off = 16; off >= 1; off >>= 1) {   // wave32 reduction
    s  += __shfl_xor(s,  off, 32);
    s2 += __shfl_xor(s2, off, 32);
  }
  float mu = s * (1.0f / Hh);
  float var = s2 * (1.0f / Hh) - mu * mu;
  float rstd = rsqrtf(var + 1e-5f);
  unsigned short* xp = xn + row * Hh;
#pragma unroll
  for (int i = 0; i < 16; ++i) {
    int idx = lane + i * 32;
    xp[idx] = f2bf((v[i] - mu) * rstd * g[idx] + b[idx]);
  }
}

// ---------- input-gate GEMM: ig[T,B,3N] = xn @ W_ih^T + b ----------
__global__ void k_ig_gemm(const unsigned short* __restrict__ xn, const unsigned short* __restrict__ W16,
                          const float* __restrict__ bias, float* __restrict__ ig) {
  int lane = threadIdx.x & 31, wave = threadIdx.x >> 5;
  int mt = blockIdx.x;
  int ntb = (blockIdx.y * 4 + wave) * 4;      // 0,4,...,44
  long arow = ((long)mt * 16 + (lane & 15)) * Hh;
  v8f acc[4] = {v8zero(), v8zero(), v8zero(), v8zero()};
#pragma unroll
  for (int kc = 0; kc < Hh / 32; ++kc) {
    v16bf a = load_a_b16(xn, arow, kc * 32, lane);
#pragma unroll
    for (int j = 0; j < 4; ++j) {
      long n = (long)(ntb + j) * 16 + (lane & 15);
      acc[j] = wmma_bf16(a, load_b_b16(W16, n, Hh, kc * 32, lane), acc[j]);
    }
  }
  int half = lane >> 4;
#pragma unroll
  for (int j = 0; j < 4; ++j) {
    long n = (long)(ntb + j) * 16 + (lane & 15);
    float bv = bias[n];
#pragma unroll
    for (int r = 0; r < 8; ++r)
      ig[((long)mt * 16 + r + 8 * half) * G3 + n] = acc[j][r] + bv;
  }
}

// ---------- persistent GRU scan ----------
// One WG, 24 waves. Per step: async-copy next ig block to LDS (ASYNCcnt),
// recurrent WMMA from LDS h (bf16), gates from LDS, two barriers.
__global__ void k_gru_scan(const float* __restrict__ ig, const unsigned short* __restrict__ Whh16,
                           const float* __restrict__ bn, unsigned short* __restrict__ gout) {
  extern __shared__ char smem[];
  float*          hF  = (float*)(smem + LDS_HF);
  unsigned short* hB  = (unsigned short*)(smem + LDS_HB);
  float*          hg  = (float*)(smem + LDS_HG);
  float*          igb = (float*)(smem + LDS_IG);
  unsigned igb_lds = (unsigned)(unsigned long long)(smem + LDS_IG); // flat[31:0] == LDS offset

  int tid = threadIdx.x, lane = tid & 31, wave = tid >> 5, half = lane >> 4;
  for (int i = tid; i < 16 * HBS; i += 768) { hF[i] = 0.f; hB[i] = 0; }

  // prime the pipeline: step 0 ig block -> LDS buffer 0 (each wave: 4 async b128 ops)
  {
    const float* gp = ig + tid * 16;
    unsigned la = igb_lds + (unsigned)tid * 64u;
#pragma unroll
    for (int i = 0; i < 4; ++i)
      asm volatile("global_load_async_to_lds_b128 %0, %1, off"
                   :: "v"(la + i * 16u), "v"(gp + i * 4) : "memory");
  }
  __syncthreads();

  long n0 = (long)wave * 32 + (lane & 15);    // wave owns tiles 2w, 2w+1 of 48
  long n1 = n0 + 16;
  long arow = (long)(lane & 15) * HBS;

  for (int t = 0; t < Tt; ++t) {
    // issue async copy of step t+1 while step t computes
    if (t + 1 < Tt) {
      const float* gp = ig + (size_t)(t + 1) * IGSTEP + tid * 16;
      unsigned la = igb_lds + ((unsigned)(t + 1) & 1u) * (IGSTEP * 4u) + (unsigned)tid * 64u;
#pragma unroll
      for (int i = 0; i < 4; ++i)
        asm volatile("global_load_async_to_lds_b128 %0, %1, off"
                     :: "v"(la + i * 16u), "v"(gp + i * 4) : "memory");
    }

    // hg[16,768] = h[16,256] @ W_hh^T (A from LDS, B streamed from L2)
    v8f acc0 = v8zero(), acc1 = v8zero();
#pragma unroll
    for (int kc = 0; kc < Nn / 32; ++kc) {
      v16bf a  = load_a_b16(hB, arow, kc * 32, lane);
      v16bf b0 = load_b_b16(Whh16, n0, Nn, kc * 32, lane);
      v16bf b1 = load_b_b16(Whh16, n1, Nn, kc * 32, lane);
      acc0 = wmma_bf16(a, b0, acc0);
      acc1 = wmma_bf16(a, b1, acc1);
    }
#pragma unroll
    for (int r = 0; r < 8; ++r) {
      int m = r + 8 * half;
      hg[m * HGS + n0] = acc0[r];
      hg[m * HGS + n1] = acc1[r];
    }
    // async batches retire in order: <=4 outstanding means batch t has landed
    if (t + 1 < Tt) asm volatile("s_wait_asynccnt 4" ::: "memory");
    else            asm volatile("s_wait_asynccnt 0" ::: "memory");
    __syncthreads();

    const float* igt = igb + (t & 1) * IGSTEP;
    for (int idx = tid; idx < 16 * Nn; idx += 768) {
      int m = idx >> 8, j = idx & 255;
      float ir = igt[m * G3 + j], iz = igt[m * G3 + Nn + j], inn = igt[m * G3 + 2 * Nn + j];
      float hr = hg[m * HGS + j], hz = hg[m * HGS + Nn + j], hn = hg[m * HGS + 2 * Nn + j];
      float r_ = sigmoidf_(ir + hr);
      float z_ = sigmoidf_(iz + hz);
      float n_ = tanhf(inn + r_ * (hn + bn[j]));
      float hold = hF[m * HBS + j];
      float hnew = n_ + z_ * (hold - n_);
      hF[m * HBS + j] = hnew;
      hB[m * HBS + j] = f2bf(hnew);
      gout[((long)t * 16 + m) * Nn + j] = f2bf(geluf_(hnew));
    }
    __syncthreads();
  }
}

// ---------- fused output projections: h += (g@W1^T+b1) * sigmoid(g@W2^T+b2) ----------
__global__ void k_out_proj(const unsigned short* __restrict__ g16,
                           const unsigned short* __restrict__ W1, const unsigned short* __restrict__ W2,
                           const float* __restrict__ b1, const float* __restrict__ b2,
                           float* __restrict__ h) {
  int lane = threadIdx.x & 31, wave = threadIdx.x >> 5;
  int mt = blockIdx.x;
  int ntb = (blockIdx.y * 4 + wave) * 2;      // 0,2,...,30
  long arow = ((long)mt * 16 + (lane & 15)) * Nn;
  v8f accL[2] = {v8zero(), v8zero()};
  v8f accG[2] = {v8zero(), v8zero()};
#pragma unroll
  for (int kc = 0; kc < Nn / 32; ++kc) {
    v16bf a = load_a_b16(g16, arow, kc * 32, lane);
#pragma unroll
    for (int j = 0; j < 2; ++j) {
      long n = (long)(ntb + j) * 16 + (lane & 15);
      accL[j] = wmma_bf16(a, load_b_b16(W1, n, Nn, kc * 32, lane), accL[j]);
      accG[j] = wmma_bf16(a, load_b_b16(W2, n, Nn, kc * 32, lane), accG[j]);
    }
  }
  int half = lane >> 4;
#pragma unroll
  for (int j = 0; j < 2; ++j) {
    long n = (long)(ntb + j) * 16 + (lane & 15);
    float bv1 = b1[n], bv2 = b2[n];
#pragma unroll
    for (int r = 0; r < 8; ++r) {
      long idx = ((long)mt * 16 + r + 8 * half) * Hh + n;
      h[idx] += (accL[j][r] + bv1) * sigmoidf_(accG[j][r] + bv2);
    }
  }
}

// ---------- decoder: y[B,T,OUT] = h[T,B,H] @ dec_W^T + dec_b ----------
__global__ void k_decoder(const float* __restrict__ h, const unsigned short* __restrict__ W16,
                          const float* __restrict__ bias, float* __restrict__ y) {
  int lane = threadIdx.x & 31;
  int mt = blockIdx.x;
  long arow = ((long)mt * 16 + (lane & 15)) * Hh;
  v8f acc[4] = {v8zero(), v8zero(), v8zero(), v8zero()};
#pragma unroll
  for (int kc = 0; kc < Hh / 32; ++kc) {
    v16bf a = load_a_f32(h, arow, kc * 32, lane);
#pragma unroll
    for (int j = 0; j < 4; ++j) {
      long n = (long)j * 16 + (lane & 15);
      acc[j] = wmma_bf16(a, load_b_b16(W16, n, Hh, kc * 32, lane), acc[j]);
    }
  }
  int half = lane >> 4;
#pragma unroll
  for (int j = 0; j < 4; ++j) {
    long n = (long)j * 16 + (lane & 15);
    float bv = bias[n];
#pragma unroll
    for (int r = 0; r < 8; ++r) {
      int m = r + 8 * half;               // batch index b
      y[((long)m * Tt + mt) * OUTF + n] = acc[j][r] + bv;
    }
  }
}

extern "C" void kernel_launch(void* const* d_in, const int* in_sizes, int n_in,
                              void* d_out, int out_size, void* d_ws, size_t ws_size,
                              hipStream_t stream) {
  (void)in_sizes; (void)n_in; (void)out_size; (void)ws_size;
  const float* x     = (const float*)d_in[0];
  const float* encW  = (const float*)d_in[1];
  const float* encB  = (const float*)d_in[2];
  const float* Wih   = (const float*)d_in[3];
  const float* Whh   = (const float*)d_in[4];
  const float* bias3 = (const float*)d_in[5];
  const float* bn    = (const float*)d_in[6];
  const float* outW  = (const float*)d_in[7];
  const float* outB  = (const float*)d_in[8];
  const float* out2W = (const float*)d_in[9];
  const float* out2B = (const float*)d_in[10];
  const float* lnG   = (const float*)d_in[11];
  const float* lnB   = (const float*)d_in[12];
  const float* decW  = (const float*)d_in[13];
  const float* decB  = (const float*)d_in[14];
  float* y = (float*)d_out;
  char* ws = (char*)d_ws;

  size_t off = 0;
  float*          h    = (float*)(ws + off);          off += (size_t)M_ROWS * Hh * 4;
  unsigned short* xn   = (unsigned short*)(ws + off); off += (size_t)M_ROWS * Hh * 2;
  float*          ig   = (float*)(ws + off);          off += (size_t)M_ROWS * G3 * 4;
  unsigned short* g16  = (unsigned short*)(ws + off); off += (size_t)M_ROWS * Nn * 2;
  unsigned short* encW16  = (unsigned short*)(ws + off); off += (size_t)Hh * INF * 2;
  unsigned short* Wih16   = (unsigned short*)(ws + off); off += (size_t)Ll * G3 * Hh * 2;
  unsigned short* Whh16   = (unsigned short*)(ws + off); off += (size_t)Ll * G3 * Nn * 2;
  unsigned short* outW16  = (unsigned short*)(ws + off); off += (size_t)Ll * Hh * Nn * 2;
  unsigned short* out2W16 = (unsigned short*)(ws + off); off += (size_t)Ll * Hh * Nn * 2;
  unsigned short* decW16  = (unsigned short*)(ws + off); off += (size_t)OUTF * Hh * 2;

  auto cvt = [&](const float* s, unsigned short* d, int n) {
    k_cvt<<<dim3((n + 255) / 256), dim3(256), 0, stream>>>(s, d, n);
  };
  cvt(encW,  encW16,  Hh * INF);
  cvt(Wih,   Wih16,   Ll * G3 * Hh);
  cvt(Whh,   Whh16,   Ll * G3 * Nn);
  cvt(outW,  outW16,  Ll * Hh * Nn);
  cvt(out2W, out2W16, Ll * Hh * Nn);
  cvt(decW,  decW16,  OUTF * Hh);

  k_encoder<<<dim3(Tt, 2), dim3(128), 0, stream>>>(x, encW16, encB, h);

  for (int l = 0; l < Ll; ++l) {
    k_layernorm<<<dim3((int)(M_ROWS / 8)), dim3(256), 0, stream>>>(h, lnG + l * Hh, lnB + l * Hh, xn);
    k_ig_gemm<<<dim3(Tt, 3), dim3(128), 0, stream>>>(
        xn, Wih16 + (size_t)l * G3 * Hh, bias3 + l * G3, ig);
    k_gru_scan<<<dim3(1), dim3(768), LDS_TOT, stream>>>(
        ig, Whh16 + (size_t)l * G3 * Nn, bn + l * Nn, g16);
    k_out_proj<<<dim3(Tt, 4), dim3(128), 0, stream>>>(
        g16, outW16 + (size_t)l * Hh * Nn, out2W16 + (size_t)l * Hh * Nn,
        outB + l * Hh, out2B + l * Hh, h);
  }

  k_decoder<<<dim3(Tt), dim3(32), 0, stream>>>(h, decW16, decB, y);
}
